// skipgram_48309792145837
// MI455X (gfx1250) — matrix operand order repaired and checked by reading, not compiled, not run on hardware
//
#include <hip/hip_runtime.h>

// ---------------------------------------------------------------------------
// Skip-gram negative-sampling loss for MI455X (gfx1250), wave32 + WMMA bf16.
//
// Per 16-row batch tile (one wave):
//   S_pos = diag( U(16x128) x Vpos^T(128x16) )   -> 4x v_wmma_f32_16x16x32_bf16
//   S_neg = diag( U(16x128) x Vsum^T(128x16) )   -> 4x v_wmma_f32_16x16x32_bf16
// where Vsum[b] = sum_k v_weight[v_neg[b,k]]  (linearity of the dot product).
// loss contribution = logsig(S_pos) + logsig(-S_neg); reduced wave-wide,
// per-tile partials land in d_ws, tiny second kernel finishes the mean.
// ---------------------------------------------------------------------------

typedef __attribute__((ext_vector_type(16))) __bf16 v16bf;
typedef __attribute__((ext_vector_type(8)))  float  v8f;

#define EMBED_DIM 128
#define N_NEG     10
#define BATCH     65536
#define TILE      16
#define WAVES_PB  8
#define THREADS   (WAVES_PB * 32)
#define NTILES    (BATCH / TILE)          // 4096
#define NBLOCKS   (NTILES / WAVES_PB)     // 512

__device__ __forceinline__ float log_sigmoid(float x) {
    // logsig(x) = min(x,0) - log1p(exp(-|x|))   (numerically stable)
    return fminf(x, 0.0f) - log1pf(__expf(-fabsf(x)));
}

__global__ __launch_bounds__(THREADS)
void sg_loss_kernel(const int*   __restrict__ u_pos,
                    const int*   __restrict__ v_pos,
                    const int*   __restrict__ v_neg,
                    const float* __restrict__ u_w,
                    const float* __restrict__ v_w,
                    float*       __restrict__ partial)
{
    const int wave = threadIdx.x >> 5;
    const int lane = threadIdx.x & 31;
    const int tile = blockIdx.x * WAVES_PB + wave;
    const int row  = lane & 15;          // tile-local row (A) / column (B)
    const int half = lane >> 4;          // which lane-half of the wave
    const int b    = tile * TILE + row;  // global batch element for this lane

    const float* __restrict__ urow = u_w + (long long)u_pos[b] * EMBED_DIM;
    const float* __restrict__ vrow = v_w + (long long)v_pos[b] * EMBED_DIM;
    const float* nrow[N_NEG];
#pragma unroll
    for (int k = 0; k < N_NEG; ++k)
        nrow[k] = v_w + (long long)v_neg[b * N_NEG + k] * EMBED_DIM;

    v8f cpos = {};
    v8f cneg = {};

#pragma unroll
    for (int q = 0; q < 4; ++q) {         // four K-chunks of 32 cover D=128
        const int kbase = q * 32;

        // ---- A (U tile): lane(half) holds chunk-K half*16 + {0..15}
        const int oa = kbase + half * 16;
        float4 a0 = *(const float4*)(urow + oa);
        float4 a1 = *(const float4*)(urow + oa + 4);
        float4 a2 = *(const float4*)(urow + oa + 8);
        float4 a3 = *(const float4*)(urow + oa + 12);
        float af[16] = { a0.x,a0.y,a0.z,a0.w, a1.x,a1.y,a1.z,a1.w,
                         a2.x,a2.y,a2.z,a2.w, a3.x,a3.y,a3.z,a3.w };

        // ---- B (Vpos^T): lane n holds column n, chunk-K half*16 + {0..15}
        const int ob = kbase + half * 16;
        float4 p0 = *(const float4*)(vrow + ob);
        float4 p1 = *(const float4*)(vrow + ob + 4);
        float4 p2 = *(const float4*)(vrow + ob + 8);
        float4 p3 = *(const float4*)(vrow + ob + 12);
        float pf[16] = { p0.x,p0.y,p0.z,p0.w, p1.x,p1.y,p1.z,p1.w,
                         p2.x,p2.y,p2.z,p2.w, p3.x,p3.y,p3.z,p3.w };

        // ---- B (Vsum^T): f32 sum of the 10 negative rows (same slice);
        //      peel k=0 into the initializer so no add-with-zero is emitted.
        float4 s0 = *(const float4*)(nrow[0] + ob);
        float4 s1 = *(const float4*)(nrow[0] + ob + 4);
        float4 s2 = *(const float4*)(nrow[0] + ob + 8);
        float4 s3 = *(const float4*)(nrow[0] + ob + 12);
        float nf[16] = { s0.x,s0.y,s0.z,s0.w, s1.x,s1.y,s1.z,s1.w,
                         s2.x,s2.y,s2.z,s2.w, s3.x,s3.y,s3.z,s3.w };
#pragma unroll
        for (int k = 1; k < N_NEG; ++k) {
            const float* p = nrow[k] + ob;
            float4 t0 = *(const float4*)(p);
            float4 t1 = *(const float4*)(p + 4);
            float4 t2 = *(const float4*)(p + 8);
            float4 t3 = *(const float4*)(p + 12);
            nf[0]+=t0.x;  nf[1]+=t0.y;  nf[2]+=t0.z;  nf[3]+=t0.w;
            nf[4]+=t1.x;  nf[5]+=t1.y;  nf[6]+=t1.z;  nf[7]+=t1.w;
            nf[8]+=t2.x;  nf[9]+=t2.y;  nf[10]+=t2.z; nf[11]+=t2.w;
            nf[12]+=t3.x; nf[13]+=t3.y; nf[14]+=t3.z; nf[15]+=t3.w;
        }

        // ---- f32 -> bf16 register tiles
        v16bf a, bp, bn;
#pragma unroll
        for (int i = 0; i < 16; ++i) {
            a[i]  = (__bf16)af[i];
            bp[i] = (__bf16)pf[i];
            bn[i] = (__bf16)nf[i];
        }

        // ---- D = A x B + C, f32 accumulate
        cpos = __builtin_amdgcn_wmma_f32_16x16x32_bf16(
                   false, a, false, bp, (short)0, cpos, false, false);
        cneg = __builtin_amdgcn_wmma_f32_16x16x32_bf16(
                   false, a, false, bn, (short)0, cneg, false, false);
    }

    // ---- diagonal extraction (C/D layout: vgpr j = {M=j | M=j+8}, N = lane%16)
    // diag(m,m): m<8 -> lane m, vgpr m ; m>=8 -> lane m+16, vgpr m-8
    const bool active = (lane < 8) || (lane >= 24);
    const int  ci     = active ? ((lane < 8) ? lane : lane - 24) : 0;
    float dpos = cpos[0], dneg = cneg[0];
#pragma unroll
    for (int j = 1; j < 8; ++j) {
        dpos = (ci == j) ? cpos[j] : dpos;
        dneg = (ci == j) ? cneg[j] : dneg;
    }

    float contrib = active ? (log_sigmoid(dpos) + log_sigmoid(-dneg)) : 0.0f;

    // ---- wave32 reduction, one partial per tile
#pragma unroll
    for (int off = 16; off > 0; off >>= 1)
        contrib += __shfl_down(contrib, off, 32);
    if (lane == 0)
        partial[tile] = contrib;
}

__global__ __launch_bounds__(256)
void sg_reduce_kernel(const float* __restrict__ partial, float* __restrict__ out)
{
    __shared__ float sm[256];
    float s = 0.0f;
    for (int i = threadIdx.x; i < NTILES; i += 256)
        s += partial[i];
    sm[threadIdx.x] = s;
    __syncthreads();
#pragma unroll
    for (int stride = 128; stride > 0; stride >>= 1) {
        if ((int)threadIdx.x < stride)
            sm[threadIdx.x] += sm[threadIdx.x + stride];
        __syncthreads();
    }
    if (threadIdx.x == 0)
        out[0] = -sm[0] / (float)BATCH;
}

extern "C" void kernel_launch(void* const* d_in, const int* in_sizes, int n_in,
                              void* d_out, int out_size, void* d_ws, size_t ws_size,
                              hipStream_t stream)
{
    const int*   u_pos = (const int*)  d_in[0];
    const int*   v_pos = (const int*)  d_in[1];
    const int*   v_neg = (const int*)  d_in[2];
    const float* u_w   = (const float*)d_in[3];
    const float* v_w   = (const float*)d_in[4];
    float*       out   = (float*)d_out;
    float*       part  = (float*)d_ws;      // NTILES floats = 16 KB scratch

    sg_loss_kernel<<<NBLOCKS, THREADS, 0, stream>>>(u_pos, v_pos, v_neg,
                                                    u_w, v_w, part);
    sg_reduce_kernel<<<1, 256, 0, stream>>>(part, out);
}
